// STHQLayer_24223615550302
// MI455X (gfx1250) — matrix operand order repaired and checked by reading, not compiled, not run on hardware
//
#include <hip/hip_runtime.h>
#include <hip/hip_bf16.h>

// ---------------- problem constants ----------------
#define BB 32
#define NN 4096
#define DD 256
#define KE 48            // 32 temporal + 16 var hyperedges

// ---------------- workspace layout (bytes) ----------------
// [denom][h_edge_raw][hedgeFrag]  <- zero-initialized region
// then wdist, waT, qT, wcatFrag, bcat
static constexpr size_t OFF_DENOM  = 0;                       // 32*48*4        = 6144
static constexpr size_t OFF_HERAW  = 6144;                    // 32*48*256*4    = 1572864
static constexpr size_t OFF_HEDGEF = 6144 + 1572864;          // 32*16384*2     = 1048576
static constexpr size_t ZERO_BYTES = OFF_HEDGEF + 1048576;    // 2627584
static constexpr size_t OFF_WDIST  = ZERO_BYTES;              // 32*4096*48*2   = 12582912
static constexpr size_t OFF_WAT    = OFF_WDIST + 12582912;    // 32*48*4096*2   = 12582912
static constexpr size_t OFF_QT     = OFF_WAT   + 12582912;    // 32*256*4096*2  = 67108864
static constexpr size_t OFF_WCATF  = OFF_QT    + 67108864;    // 512*256*2      = 262144
static constexpr size_t OFF_BCAT   = OFF_WCATF + 262144;      // 256*4         = 1024

// ---------------- WMMA types ----------------
typedef __bf16 v16bf __attribute__((ext_vector_type(16)));
typedef float  v8f   __attribute__((ext_vector_type(8)));

union AF { v16bf v; uint4 q[2]; };
union CF { v8f v; float f[8]; };

__device__ __forceinline__ v8f wmma_bf16(v16bf a, v16bf b, v8f c) {
    return __builtin_amdgcn_wmma_f32_16x16x32_bf16(false, a, false, b, (short)0, c, false, false);
}

__device__ __forceinline__ unsigned short f2bf(float f) {
    union { float f; unsigned u; } x; x.f = f;
    unsigned u = x.u;
    unsigned r = (u + 0x7FFFu + ((u >> 16) & 1u)) >> 16;
    return (unsigned short)r;
}
__device__ __forceinline__ float bf2f(unsigned short h) {
    union { unsigned u; float f; } x; x.u = ((unsigned)h) << 16;
    return x.f;
}

// A fragment (16x32 bf16) from a row-major [M][K] matrix (LDS or global).
// Contiguous k order used here: lane<16 -> M=lane, K = k0+[0..15];
//             lane>=16 -> M=lane-16, K = k0+[16..31].
__device__ __forceinline__ v16bf afrag_rm(const unsigned short* base, int stride,
                                          int m0, int k0, int lane) {
    int half = lane >> 4, l = lane & 15;
    const unsigned short* p = base + (size_t)(m0 + l) * stride + k0 + half * 16;
    AF a;
    a.q[0] = *reinterpret_cast<const uint4*>(p);
    a.q[1] = *reinterpret_cast<const uint4*>(p + 8);
    return a.v;
}

// B fragment (32x16 bf16) from a packed fragment buffer:
// frag idx -> 32 lanes x 16 bf16 contiguous; lane<16: N=lane, K=k0+slot;
// lane>=16: N=lane-16, K=k0+16+slot.
__device__ __forceinline__ v16bf bfrag_packed(const unsigned short* frags, int fragIdx, int lane) {
    const unsigned short* p = frags + ((size_t)fragIdx * 32 + lane) * 16;
    AF a;
    a.q[0] = *reinterpret_cast<const uint4*>(p);
    a.q[1] = *reinterpret_cast<const uint4*>(p + 8);
    return a.v;
}

// B fragment from a column-major source stored as B^T row-major [N][K].
__device__ __forceinline__ v16bf bfrag_colmajor(const unsigned short* base, int stride,
                                                int n0, int k0, int lane) {
    int half = lane >> 4, l = lane & 15;
    const unsigned short* p = base + (size_t)(n0 + l) * stride + k0 + half * 16;
    AF a;
    a.q[0] = *reinterpret_cast<const uint4*>(p);
    a.q[1] = *reinterpret_cast<const uint4*>(p + 8);
    return a.v;
}

// Quaternion kron-expanded weight W[r][c] from 64x64 blocks R,I,J,K.
// sel = (r>>6)^(c>>6); sign from quaternion multiplication table.
__device__ __forceinline__ float qkron_w(const float* R, const float* I,
                                         const float* J, const float* K, int r, int c) {
    int br = r >> 6, bc = c >> 6, rr = r & 63, cc = c & 63;
    int sel = br ^ bc;
    const float* M = (sel == 0) ? R : (sel == 1) ? I : (sel == 2) ? J : K;
    float v = M[rr * 64 + cc];
    // negatives at (br,bc) in {(0,1),(0,2),(0,3),(1,2),(2,3),(3,1)} -> mask 10318
    return ((10318 >> (br * 4 + bc)) & 1) ? -v : v;
}

// ================= K0: zero accumulators =================
__global__ void k0_init(float* p, size_t nfloats) {
    size_t i = (size_t)blockIdx.x * blockDim.x + threadIdx.x;
    size_t stride = (size_t)gridDim.x * blockDim.x;
    for (; i < nfloats; i += stride) p[i] = 0.0f;
}

// ================= K1: per-cell kernel weights =================
__global__ __launch_bounds__(256) void k1_cellw(
    const float* spike, const float* tn, const float* mask, const int* vid,
    const float* tau, const float* omega_log, const float* vaff,
    unsigned short* wdist, unsigned short* waT, float* denom) {
    int b = blockIdx.x >> 4;
    int n = ((blockIdx.x & 15) << 8) + threadIdx.x;
    size_t cell = (size_t)b * NN + n;
    float t = tn[cell], s = spike[cell], mk = mask[cell];
    int v = vid[cell];

    float kt[32]; float st = 0.f;
    #pragma unroll
    for (int j = 0; j < 32; ++j) {
        float om = fmaxf(__expf(omega_log[j]), 1e-3f);
        float df = (t - tau[j]) / om;
        float e = __expf(-0.5f * df * df);
        kt[j] = e; st += e;
    }
    float kv[16]; float mx = -1e30f;
    #pragma unroll
    for (int j = 0; j < 16; ++j) { kv[j] = vaff[v * 16 + j]; mx = fmaxf(mx, kv[j]); }
    float sv = 0.f;
    #pragma unroll
    for (int j = 0; j < 16; ++j) { kv[j] = __expf(kv[j] - mx); sv += kv[j]; }
    float inv_sv = 1.0f / sv;
    #pragma unroll
    for (int j = 0; j < 16; ++j) kv[j] *= inv_sv;

    float dt = fmaxf(st * mk, 1e-6f);
    float dv = fmaxf(mk, 1e-6f);   // softmax sums to 1

    __shared__ float red[KE];
    if (threadIdx.x < KE) red[threadIdx.x] = 0.f;
    __syncthreads();

    unsigned short* wrow = wdist + cell * KE;
    unsigned short* wab = waT + (size_t)b * KE * NN;
    #pragma unroll
    for (int j = 0; j < 32; ++j) {
        wrow[j] = f2bf(kt[j] * mk / dt);
        float wa = s * kt[j] * mk;
        wab[(size_t)j * NN + n] = f2bf(wa);
        atomicAdd(&red[j], wa);
    }
    #pragma unroll
    for (int j = 0; j < 16; ++j) {
        wrow[32 + j] = f2bf(kv[j] * mk / dv);
        float wa = s * kv[j] * mk;
        wab[(size_t)(32 + j) * NN + n] = f2bf(wa);
        atomicAdd(&red[32 + j], wa);
    }
    __syncthreads();
    if (threadIdx.x < KE) atomicAdd(&denom[b * KE + threadIdx.x], red[threadIdx.x]);
}

// ================= K2: q fp32 -> qT bf16 [B][256][4096] =================
__global__ __launch_bounds__(256) void k2_transpose(const float* q, unsigned short* qT) {
    __shared__ __align__(16) unsigned short tile[64][72];
    int blk = blockIdx.x;
    int b = blk >> 8;
    int rem = blk & 255;
    int ntile = rem >> 2;       // 0..63 -> n block of 64
    int dtile = rem & 3;        // 0..3  -> d block of 64
    int c = threadIdx.x & 63;
    int r0 = threadIdx.x >> 6;
    #pragma unroll
    for (int rr = 0; rr < 64; rr += 4) {
        int r = rr + r0;
        tile[r][c] = f2bf(q[(((size_t)b * NN) + ntile * 64 + r) * DD + dtile * 64 + c]);
    }
    __syncthreads();
    #pragma unroll
    for (int rr = 0; rr < 64; rr += 4) {
        int r = rr + r0; // d offset
        qT[(((size_t)b * DD) + dtile * 64 + r) * NN + ntile * 64 + c] = tile[c][r];
    }
}

// ================= K3: build Wcat (B-frag packed) + fused bias =================
__global__ __launch_bounds__(256) void k3_build(
    const float* alpha_logit,
    const float* mphR, const float* mphI, const float* mphJ, const float* mphK,
    const float* mph_b, const float* mplW, const float* mpl_b,
    unsigned short* wcatF, float* bcat) {
    int idx = blockIdx.x * blockDim.x + threadIdx.x;
    float al = 1.0f / (1.0f + __expf(-alpha_logit[0]));
    if (idx < 131072) {
        int s  = idx & 15;
        int l  = (idx >> 4) & 31;
        int nt = (idx >> 9) & 15;
        int kt = idx >> 13;
        int kin  = kt * 32 + ((l & 16) ? 16 + s : s);
        int dout = nt * 16 + (l & 15);
        float wv;
        if (kin < 256) wv = al * qkron_w(mphR, mphI, mphJ, mphK, dout, kin);
        else           wv = (1.0f - al) * mplW[dout * 256 + (kin - 256)];
        wcatF[idx] = f2bf(wv);
    }
    if (idx < 256) bcat[idx] = al * mph_b[idx] + (1.0f - al) * mpl_b[idx];
}

// ================= K5: aggregation GEMM (h_edge_raw += waT x q) =================
__global__ __launch_bounds__(256) void k5_aggr(const unsigned short* waT,
                                               const unsigned short* qT, float* heraw) {
    int b = blockIdx.x >> 3;
    int nb0 = (blockIdx.x & 7) * 512;
    int lane = threadIdx.x & 31;
    int w = threadIdx.x >> 5;
    int ntb = w * 2;

    CF acc[6];
    #pragma unroll
    for (int i = 0; i < 6; ++i)
        #pragma unroll
        for (int j = 0; j < 8; ++j) acc[i].f[j] = 0.f;

    const unsigned short* wab = waT + (size_t)b * KE * NN;
    const unsigned short* qtb = qT + (size_t)b * DD * NN;
    for (int ki = 0; ki < 16; ++ki) {
        int n0 = nb0 + ki * 32;
        v16bf a0 = afrag_rm(wab, NN, 0,  n0, lane);
        v16bf a1 = afrag_rm(wab, NN, 16, n0, lane);
        v16bf a2 = afrag_rm(wab, NN, 32, n0, lane);
        v16bf b0 = bfrag_colmajor(qtb, NN, (ntb + 0) * 16, n0, lane);
        v16bf b1 = bfrag_colmajor(qtb, NN, (ntb + 1) * 16, n0, lane);
        acc[0].v = wmma_bf16(a0, b0, acc[0].v);
        acc[1].v = wmma_bf16(a0, b1, acc[1].v);
        acc[2].v = wmma_bf16(a1, b0, acc[2].v);
        acc[3].v = wmma_bf16(a1, b1, acc[3].v);
        acc[4].v = wmma_bf16(a2, b0, acc[4].v);
        acc[5].v = wmma_bf16(a2, b1, acc[5].v);
    }
    int halfId = lane >> 4, cL = lane & 15;
    #pragma unroll
    for (int mt = 0; mt < 3; ++mt)
        #pragma unroll
        for (int nti = 0; nti < 2; ++nti)
            #pragma unroll
            for (int j = 0; j < 8; ++j) {
                int k = mt * 16 + j + halfId * 8;
                int d = (ntb + nti) * 16 + cL;
                atomicAdd(&heraw[((size_t)b * KE + k) * DD + d], acc[mt * 2 + nti].f[j]);
            }
}

// ================= K4: edge qlinear -> hedgeFrag (B-frag packed, K padded 64) ====
__global__ __launch_bounds__(256) void k4_edge(
    const float* heraw, const float* denom,
    const float* eptR, const float* eptI, const float* eptJ, const float* eptK, const float* eptb,
    const float* epvR, const float* epvI, const float* epvJ, const float* epvK, const float* epvb,
    unsigned short* hedgeF) {
    int b = blockIdx.x / KE;
    int k = blockIdx.x % KE;
    bool isT = (k < 32);
    const float* R  = isT ? eptR : epvR;
    const float* I  = isT ? eptI : epvI;
    const float* J  = isT ? eptJ : epvJ;
    const float* K  = isT ? eptK : epvK;
    const float* bb = isT ? eptb : epvb;

    __shared__ float hin[DD];
    int d = threadIdx.x;
    float dn = fmaxf(denom[b * KE + k], 1e-6f);
    hin[d] = heraw[((size_t)b * KE + k) * DD + d] / dn;
    __syncthreads();

    float acc = bb[d];
    for (int c = 0; c < DD; ++c) acc += qkron_w(R, I, J, K, d, c) * hin[c];

    // frag position for (K-dim = k, N-dim = d), K padded to 64 (pad zeroed in K0)
    int kt5 = k >> 5, kk32 = k & 31;
    int lane = (d & 15) + ((kk32 & 16) ? 16 : 0);
    int slot = kk32 & 15;
    int nt = d >> 4;
    hedgeF[(size_t)b * 16384 + ((size_t)(kt5 * 16 + nt) * 32 + lane) * 16 + slot] = f2bf(acc);
}

// ================= K6: fused dist-einsum + hamilton + GEMM + LN ================
__global__ __launch_bounds__(256) void k6_main(
    const float* q, const float* mask,
    const unsigned short* wdist_g, const unsigned short* hedgeF_g,
    const unsigned short* wcatF, const float* bcat,
    const float* lnrw, const float* lnrb, const float* lniw, const float* lnib,
    const float* lnjw, const float* lnjb, const float* lnkw, const float* lnkb,
    float* out) {
    __shared__ __align__(16) unsigned short sWD[64 * 64];   // [row][k] (k padded to 64)
    __shared__ __align__(16) unsigned short sX2[64 * 520];  // [row][0..255 msg_h | 256..511 h] + pad

    int b = blockIdx.x >> 6;
    int n0 = (blockIdx.x & 63) * 64;
    int tid = threadIdx.x;
    int lane = tid & 31;
    int w = tid >> 5;

    // ---- phase A: stage dist weights into LDS ----
    {
        int row = tid >> 2, part = tid & 3;
        const unsigned short* src = wdist_g + ((size_t)(b * NN + n0 + row)) * KE;
        if (part < 3) {
            *reinterpret_cast<uint4*>(&sWD[row * 64 + part * 16])     = *reinterpret_cast<const uint4*>(src + part * 16);
            *reinterpret_cast<uint4*>(&sWD[row * 64 + part * 16 + 8]) = *reinterpret_cast<const uint4*>(src + part * 16 + 8);
        } else {
            uint4 z; z.x = z.y = z.z = z.w = 0u;
            *reinterpret_cast<uint4*>(&sWD[row * 64 + 48]) = z;
            *reinterpret_cast<uint4*>(&sWD[row * 64 + 56]) = z;
        }
    }
    __syncthreads();

    int Mt = w >> 1;            // 0..3 (16-row tile)
    int ntb = (w & 1) * 8;      // 0 or 8 (8 N-tiles)
    int halfId = lane >> 4, cL = lane & 15;

    // ---- phase B: h_per_cell = wdist (64x64) x hedge (64x256) via WMMA ----
    {
        CF acc[8];
        #pragma unroll
        for (int i = 0; i < 8; ++i)
            #pragma unroll
            for (int j = 0; j < 8; ++j) acc[i].f[j] = 0.f;
        const unsigned short* hb = hedgeF_g + (size_t)b * 16384;
        #pragma unroll
        for (int kt = 0; kt < 2; ++kt) {
            v16bf a = afrag_rm(sWD, 64, Mt * 16, kt * 32, lane);
            #pragma unroll
            for (int nti = 0; nti < 8; ++nti) {
                v16bf bb = bfrag_packed(hb, kt * 16 + ntb + nti, lane);
                acc[nti].v = wmma_bf16(a, bb, acc[nti].v);
            }
        }
        int rbase = Mt * 16 + halfId * 8;
        #pragma unroll
        for (int nti = 0; nti < 8; ++nti)
            #pragma unroll
            for (int j = 0; j < 8; ++j)
                sX2[(rbase + j) * 520 + 256 + (ntb + nti) * 16 + cL] = f2bf(acc[nti].f[j]);
    }
    __syncthreads();

    // ---- phase C: hamilton(h_per_cell, q) -> left half of X2 ----
    {
        int row = tid >> 2;
        int ub = (tid & 3) * 16;
        const float* qr = q + ((size_t)(b * NN + n0 + row)) * DD;
        const unsigned short* hrow = &sX2[row * 520 + 256];
        #pragma unroll
        for (int i = 0; i < 16; ++i) {
            int u = ub + i;
            float hr = bf2f(hrow[u]);
            float hi = bf2f(hrow[64 + u]);
            float hj = bf2f(hrow[128 + u]);
            float hk = bf2f(hrow[192 + u]);
            float q0 = qr[u], q1 = qr[64 + u], q2 = qr[128 + u], q3 = qr[192 + u];
            sX2[row * 520 + u]       = f2bf(hr * q0 - hi * q1 - hj * q2 - hk * q3);
            sX2[row * 520 + 64 + u]  = f2bf(hr * q1 + hi * q0 + hj * q3 - hk * q2);
            sX2[row * 520 + 128 + u] = f2bf(hr * q2 - hi * q3 + hj * q0 + hk * q1);
            sX2[row * 520 + 192 + u] = f2bf(hr * q3 + hi * q2 - hj * q1 + hk * q0);
        }
    }
    __syncthreads();

    // ---- phase D: msg = X2 (64x512) x Wcat^T via WMMA ----
    CF macc[8];
    #pragma unroll
    for (int i = 0; i < 8; ++i)
        #pragma unroll
        for (int j = 0; j < 8; ++j) macc[i].f[j] = 0.f;
    for (int kt = 0; kt < 16; ++kt) {
        v16bf a = afrag_rm(sX2, 520, Mt * 16, kt * 32, lane);
        #pragma unroll
        for (int nti = 0; nti < 8; ++nti) {
            v16bf bb = bfrag_packed(wcatF, kt * 16 + ntb + nti, lane);
            macc[nti].v = wmma_bf16(a, bb, macc[nti].v);
        }
    }

    // ---- phase E: bias + residual + quaternion LayerNorm + mask + store ----
    const float* LNW[4] = {lnrw, lniw, lnjw, lnkw};
    const float* LNB[4] = {lnrb, lnib, lnjb, lnkb};
    int qq0 = ntb >> 2;  // 0 (w even) or 2 (w odd); nti>=4 -> +1
    const float* W0 = LNW[qq0], * B0 = LNB[qq0];
    const float* W1 = LNW[qq0 + 1], * B1 = LNB[qq0 + 1];

    #pragma unroll
    for (int j = 0; j < 8; ++j) {
        int row = Mt * 16 + j + halfId * 8;
        int n = n0 + row;
        const float* qrow = q + ((size_t)(b * NN) + n) * DD;
        float mk = mask[(size_t)b * NN + n];
        float xs[8];
        float s0 = 0.f, s1 = 0.f, ss0 = 0.f, ss1 = 0.f;
        #pragma unroll
        for (int nti = 0; nti < 8; ++nti) {
            int col = (ntb + nti) * 16 + cL;
            float x = qrow[col] + macc[nti].f[j] + bcat[col];
            xs[nti] = x;
            if (nti < 4) { s0 += x; ss0 += x * x; } else { s1 += x; ss1 += x * x; }
        }
        // reduce across the 16 lanes of this half-wave (one quaternion component = 64 ch)
        #pragma unroll
        for (int m = 1; m < 16; m <<= 1) {
            s0 += __shfl_xor(s0, m, 32);
            ss0 += __shfl_xor(ss0, m, 32);
            s1 += __shfl_xor(s1, m, 32);
            ss1 += __shfl_xor(ss1, m, 32);
        }
        float mu0 = s0 * (1.f / 64.f), var0 = ss0 * (1.f / 64.f) - mu0 * mu0;
        float mu1 = s1 * (1.f / 64.f), var1 = ss1 * (1.f / 64.f) - mu1 * mu1;
        float rs0 = rsqrtf(var0 + 1e-5f), rs1 = rsqrtf(var1 + 1e-5f);

        float* orow = out + ((size_t)(b * NN) + n) * DD;
        #pragma unroll
        for (int nti = 0; nti < 8; ++nti) {
            int col = (ntb + nti) * 16 + cL;
            int cc = col & 63;
            float mu = (nti < 4) ? mu0 : mu1;
            float rs = (nti < 4) ? rs0 : rs1;
            const float* Wv = (nti < 4) ? W0 : W1;
            const float* Bv = (nti < 4) ? B0 : B1;
            float y = (xs[nti] - mu) * rs * Wv[cc] + Bv[cc];
            orow[col] = y * mk;
        }
    }
}

// ================= host =================
extern "C" void kernel_launch(void* const* d_in, const int* in_sizes, int n_in,
                              void* d_out, int out_size, void* d_ws, size_t ws_size,
                              hipStream_t stream) {
    const float* q     = (const float*)d_in[0];
    const float* spike = (const float*)d_in[1];
    const float* tn    = (const float*)d_in[2];
    const float* mask  = (const float*)d_in[3];
    const int*   vid   = (const int*)d_in[4];
    const float* tau   = (const float*)d_in[5];
    const float* omega = (const float*)d_in[6];
    const float* vaff  = (const float*)d_in[7];
    const float* alog  = (const float*)d_in[8];
    const float* eptR = (const float*)d_in[9],  *eptI = (const float*)d_in[10];
    const float* eptJ = (const float*)d_in[11], *eptK = (const float*)d_in[12];
    const float* eptb = (const float*)d_in[13];
    const float* epvR = (const float*)d_in[14], *epvI = (const float*)d_in[15];
    const float* epvJ = (const float*)d_in[16], *epvK = (const float*)d_in[17];
    const float* epvb = (const float*)d_in[18];
    const float* mphR = (const float*)d_in[19], *mphI = (const float*)d_in[20];
    const float* mphJ = (const float*)d_in[21], *mphK = (const float*)d_in[22];
    const float* mphb = (const float*)d_in[23];
    const float* mplW = (const float*)d_in[24], *mplb = (const float*)d_in[25];
    const float* lnrw = (const float*)d_in[26], *lnrb = (const float*)d_in[27];
    const float* lniw = (const float*)d_in[28], *lnib = (const float*)d_in[29];
    const float* lnjw = (const float*)d_in[30], *lnjb = (const float*)d_in[31];
    const float* lnkw = (const float*)d_in[32], *lnkb = (const float*)d_in[33];

    char* ws = (char*)d_ws;
    float*          denom  = (float*)(ws + OFF_DENOM);
    float*          heraw  = (float*)(ws + OFF_HERAW);
    unsigned short* hedgeF = (unsigned short*)(ws + OFF_HEDGEF);
    unsigned short* wdist  = (unsigned short*)(ws + OFF_WDIST);
    unsigned short* waT    = (unsigned short*)(ws + OFF_WAT);
    unsigned short* qT     = (unsigned short*)(ws + OFF_QT);
    unsigned short* wcatF  = (unsigned short*)(ws + OFF_WCATF);
    float*          bcat   = (float*)(ws + OFF_BCAT);

    // K0: zero denom + h_edge_raw + hedgeFrag (pad slots)
    k0_init<<<2566, 256, 0, stream>>>((float*)(ws + OFF_DENOM), ZERO_BYTES / 4);
    // K1: per-cell weights + denominators
    k1_cellw<<<BB * (NN / 256), 256, 0, stream>>>(spike, tn, mask, vid, tau, omega, vaff,
                                                  wdist, waT, denom);
    // K2: q -> bf16 transpose
    k2_transpose<<<BB * 64 * 4, 256, 0, stream>>>(q, qT);
    // K3: Wcat fragment pack + fused bias
    k3_build<<<512, 256, 0, stream>>>(alog, mphR, mphI, mphJ, mphK, mphb, mplW, mplb,
                                      wcatF, bcat);
    // K5: aggregation GEMM (atomic fp32 reduction over n-chunks)
    k5_aggr<<<BB * 8, 256, 0, stream>>>(waT, qT, heraw);
    // K4: edge qlinear -> fragment-packed edge matrix
    k4_edge<<<BB * KE, 256, 0, stream>>>(heraw, denom,
                                         eptR, eptI, eptJ, eptK, eptb,
                                         epvR, epvI, epvJ, epvK, epvb, hedgeF);
    // K6: fused dist-einsum + hamilton + message GEMM + LayerNorm
    k6_main<<<BB * (NN / 64), 256, 0, stream>>>(q, mask, wdist, hedgeF, wcatF, bcat,
                                                lnrw, lnrb, lniw, lnib, lnjw, lnjb,
                                                lnkw, lnkb, (float*)d_out);
}
